// NCA_73538430042281
// MI455X (gfx1250) — compile-verified, matrix operand order checked by reading, not run on hardware
//
#include <hip/hip_runtime.h>
#include <hip/hip_bf16.h>

typedef __attribute__((ext_vector_type(16))) _Float16 v16h;
typedef __attribute__((ext_vector_type(8)))  _Float16 v8h;
typedef __attribute__((ext_vector_type(8)))  float    v8f;

#define BLK 256
#define PIX 128          // pixels per block (half of one W=256 row)
#define HW  256
#define CH  16
#define WPRE_HALVES (64*64 + 32*64 + 16*32)   // 6656 halves = 13312 bytes

// LDS layout (bytes), total 54784:
//  region A: [0, 25088)    xt[3][130][16] f32 (24960B) -> reused: H1[128][64] f16 @0, D[128][16] f32 @16384
//  region B: [25088, 41472) A1[128][64] f16            -> reused: H2[128][32] f16
//  region C: [41472, 54784) w1T[64][64] f16, w2T[32][64] f16, w3T[16][32] f16

__device__ __forceinline__ v8f vzero() {
    v8f v = {0.f, 0.f, 0.f, 0.f, 0.f, 0.f, 0.f, 0.f};
    return v;
}

__device__ __forceinline__ float fast_tanh(float x) {
#if __has_builtin(__builtin_amdgcn_tanhf)
    return __builtin_amdgcn_tanhf(x);          // V_TANH_F32 (gfx1250 trans op)
#else
    // branch-free: tanh(x) = (e^{2x}-1)/(e^{2x}+1), clamp to keep exp2 finite
    float xc = fminf(fmaxf(x, -9.f), 9.f);
    float e  = __builtin_amdgcn_exp2f(xc * 2.88539008177793f);  // exp(2x)
    return (e - 1.f) * __builtin_amdgcn_rcpf(e + 1.f);
#endif
}

// 16x32 f16 WMMA A/B fragment from row-major LDS [row][K], K-stride `stride`
// halves. Lane l holds K = kb..kb+7 and kb+16..kb+23, kb = kstep+(l>>4)*8.
__device__ __forceinline__ v16h load_frag(const _Float16* base, int row, int stride, int kb) {
    const v8h lo = *(const v8h*)(base + row * stride + kb);
    const v8h hi = *(const v8h*)(base + row * stride + kb + 16);
    return __builtin_shufflevector(lo, hi, 0, 1, 2, 3, 4, 5, 6, 7, 8, 9, 10, 11, 12, 13, 14, 15);
}

// One-block prep: transpose weights to f16 [N][K] (K padded) in d_ws.
__global__ __launch_bounds__(BLK) void NCA_prep_kernel(
    const float* __restrict__ w1, const float* __restrict__ w2,
    const float* __restrict__ w3, _Float16* __restrict__ ws)
{
    const int t = threadIdx.x;
    _Float16* w1T = ws;                  // [64][64]
    _Float16* w2T = ws + 64 * 64;        // [32][64]
    _Float16* w3T = w2T + 32 * 64;       // [16][32]
    for (int i = t; i < 64 * 64; i += BLK) {
        int n = i >> 6, k = i & 63;
        w1T[i] = (k < 48) ? (_Float16)w1[k * 64 + n] : (_Float16)0.f;
    }
    for (int i = t; i < 32 * 64; i += BLK) {
        int n = i >> 6, k = i & 63;
        w2T[i] = (_Float16)w2[k * 32 + n];
    }
    for (int i = t; i < 16 * 32; i += BLK) {
        int n = i >> 5, k = i & 31;
        w3T[i] = (_Float16)w3[k * 16 + n];
    }
}

__global__ __launch_bounds__(BLK) void NCA_73538430042281_kernel(
    const float* __restrict__ x,
    const float* __restrict__ w1, const float* __restrict__ b1,
    const float* __restrict__ w2, const float* __restrict__ b2,
    const float* __restrict__ w3, const float* __restrict__ b3,
    const unsigned char* __restrict__ mask,
    const _Float16* __restrict__ wpre,    // pre-transposed weights (or nullptr)
    float* __restrict__ out)
{
    __shared__ __align__(16) unsigned char smem[54784];
    float*    xt  = (float*)smem;                         // [3][130][16]
    _Float16* H1  = (_Float16*)smem;                      // [128][64]
    float*    D   = (float*)(smem + 16384);               // [128][16]
    _Float16* A1  = (_Float16*)(smem + 25088);            // [128][64]
    _Float16* H2  = (_Float16*)(smem + 25088);            // [128][32]
    _Float16* w1T = (_Float16*)(smem + 41472);            // [64][64]
    _Float16* w2T = w1T + 64 * 64;                        // [32][64]
    _Float16* w3T = w2T + 32 * 64;                        // [16][32]

    const int tid = threadIdx.x;
    const int blk = blockIdx.x;
    const int w0  = (blk & 1) * PIX;
    const int h   = (blk >> 1) & (HW - 1);
    const int b   = blk >> 9;
    const int pix0 = blk * PIX;

    // ---- Stage weights into LDS ----
    if (wpre) {
        const uint4* src = (const uint4*)wpre;
        uint4*       dst = (uint4*)w1T;
        for (int i = tid; i < WPRE_HALVES / 8; i += BLK)   // 832 x 16B
            dst[i] = src[i];
    } else {
        for (int i = tid; i < 64 * 64; i += BLK) {
            int n = i >> 6, k = i & 63;
            w1T[i] = (k < 48) ? (_Float16)w1[k * 64 + n] : (_Float16)0.f;
        }
        for (int i = tid; i < 32 * 64; i += BLK) {
            int n = i >> 6, k = i & 63;
            w2T[i] = (_Float16)w2[k * 32 + n];
        }
        for (int i = tid; i < 16 * 32; i += BLK) {
            int n = i >> 5, k = i & 31;
            w3T[i] = (_Float16)w3[k * 16 + n];
        }
    }

    // ---- Stage x halo tile (vectorized: 4 channels / float4) ----
    for (int i = tid; i < 3 * 130 * 4; i += BLK) {
        int r   = i / (130 * 4);
        int rem = i - r * (130 * 4);
        int cc  = rem >> 2;
        int c4  = (rem & 3) * 4;
        int hh  = h + r - 1;
        int ww  = w0 + cc - 1;
        float4 v = make_float4(0.f, 0.f, 0.f, 0.f);
        if ((unsigned)hh < (unsigned)HW && (unsigned)ww < (unsigned)HW)
            v = *(const float4*)(x + (((b * HW + hh) * HW + ww) * CH) + c4);
        *(float4*)(xt + (r * 130 + cc) * 16 + c4) = v;
    }
    __syncthreads();

    // ---- Perceive -> A1 f16 [128][64] (K: c0_I,c0_dx,c0_dy,...; K 48..63 zero) ----
    {
        const int c  = tid & 15;
        const int pb = (tid >> 4) * 8;
#pragma unroll
        for (int j = 0; j < 8; ++j) {
            int p = pb + j;
            const float* col0 = xt + (0 * 130 + p) * 16 + c;
            const float* col1 = xt + (1 * 130 + p) * 16 + c;
            const float* col2 = xt + (2 * 130 + p) * 16 + c;
            float v00 = col0[0],  v01 = col0[16], v02 = col0[32];
            float v10 = col1[0],  v11 = col1[16], v12 = col1[32];
            float v20 = col2[0],  v21 = col2[16], v22 = col2[32];
            float I   = v11;
            float sdx = ((v02 - v00) + 2.f * (v12 - v10) + (v22 - v20)) * 0.125f;
            float sdy = ((v20 - v00) + 2.f * (v21 - v01) + (v22 - v02)) * 0.125f;
            A1[p * 64 + 3 * c + 0] = (_Float16)I;
            A1[p * 64 + 3 * c + 1] = (_Float16)sdx;
            A1[p * 64 + 3 * c + 2] = (_Float16)sdy;
            A1[p * 64 + 48 + c]    = (_Float16)0.f;
        }
    }
    __syncthreads();   // xt reads done (H1/D alias xt region)

    const int lane = tid & 31;
    const int wv   = tid >> 5;
    const int lm   = lane & 15;
    const int kseg = (lane >> 4) * 8;
    const int mrow = wv * 16 + lm;
    const int mb   = (lane >> 4) * 8;

    // ---- GEMM1: [16x64] x [64x64] -> tanh -> H1 ----
    v8f acc1[4] = {vzero(), vzero(), vzero(), vzero()};
#pragma unroll
    for (int ks = 0; ks < 2; ++ks) {
        v16h a = load_frag(A1, mrow, 64, ks * 32 + kseg);
#pragma unroll
        for (int n = 0; n < 4; ++n) {
            v16h bf = load_frag(w1T, n * 16 + lm, 64, ks * 32 + kseg);
            acc1[n] = __builtin_amdgcn_wmma_f32_16x16x32_f16(
                false, a, false, bf, (short)0, acc1[n], false, false);
        }
    }
#pragma unroll
    for (int n = 0; n < 4; ++n) {
        int col = n * 16 + lm;
        float bias = b1[col];
#pragma unroll
        for (int i = 0; i < 8; ++i)
            H1[(wv * 16 + mb + i) * 64 + col] = (_Float16)fast_tanh(acc1[n][i] + bias);
    }
    __syncthreads();   // A1 reads done (H2 aliases A1 region)

    // ---- GEMM2: [16x64] x [64x32] -> tanh -> H2 ----
    v8f acc2[2] = {vzero(), vzero()};
#pragma unroll
    for (int ks = 0; ks < 2; ++ks) {
        v16h a = load_frag(H1, mrow, 64, ks * 32 + kseg);
#pragma unroll
        for (int n = 0; n < 2; ++n) {
            v16h bf = load_frag(w2T, n * 16 + lm, 64, ks * 32 + kseg);
            acc2[n] = __builtin_amdgcn_wmma_f32_16x16x32_f16(
                false, a, false, bf, (short)0, acc2[n], false, false);
        }
    }
#pragma unroll
    for (int n = 0; n < 2; ++n) {
        int col = n * 16 + lm;
        float bias = b2[col];
#pragma unroll
        for (int i = 0; i < 8; ++i)
            H2[(wv * 16 + mb + i) * 32 + col] = (_Float16)fast_tanh(acc2[n][i] + bias);
    }
    __syncthreads();

    // ---- GEMM3: [16x32] x [32x16] -> D (f32) ----
    {
        v8f acc3 = vzero();
        v16h a  = load_frag(H2, mrow, 32, kseg);
        v16h bf = load_frag(w3T, lm, 32, kseg);
        acc3 = __builtin_amdgcn_wmma_f32_16x16x32_f16(
            false, a, false, bf, (short)0, acc3, false, false);
        float bias = b3[lm];
#pragma unroll
        for (int i = 0; i < 8; ++i)
            D[(wv * 16 + mb + i) * 16 + lm] = acc3[i] + bias;
    }
    __syncthreads();

    // ---- Epilogue: out = x + D*mask, float4-vectorized, fully coalesced ----
    {
        const int base = pix0 * CH;                 // float index of block start
        // thread's 8 floats (= half a pixel) never cross a pixel boundary pair:
        const float m = mask[pix0 + (tid >> 1)] ? 1.f : 0.f;
        const float4* xv  = (const float4*)(x + base);
        const float4* dv  = (const float4*)D;
        float4*       ov  = (float4*)(out + base);
#pragma unroll
        for (int q = 0; q < 2; ++q) {
            int i = tid * 2 + q;
            float4 xa = xv[i];
            float4 da = dv[i];
            ov[i] = make_float4(xa.x + da.x * m, xa.y + da.y * m,
                                xa.z + da.z * m, xa.w + da.w * m);
        }
    }
}

extern "C" void kernel_launch(void* const* d_in, const int* in_sizes, int n_in,
                              void* d_out, int out_size, void* d_ws, size_t ws_size,
                              hipStream_t stream) {
    const float* x    = (const float*)d_in[0];
    const float* w1   = (const float*)d_in[1];
    const float* b1   = (const float*)d_in[2];
    const float* w2   = (const float*)d_in[3];
    const float* b2   = (const float*)d_in[4];
    const float* w3   = (const float*)d_in[5];
    const float* b3   = (const float*)d_in[6];
    const unsigned char* mask = (const unsigned char*)d_in[7];
    float* out = (float*)d_out;

    _Float16* wpre = nullptr;
    if (ws_size >= (size_t)WPRE_HALVES * sizeof(_Float16)) {
        wpre = (_Float16*)d_ws;
        NCA_prep_kernel<<<1, BLK, 0, stream>>>(w1, w2, w3, wpre);
    }

    const int nblocks = 16 * 256 * (256 / PIX);   // 8192
    NCA_73538430042281_kernel<<<nblocks, BLK, 0, stream>>>(
        x, w1, b1, w2, b2, w3, b3, mask, wpre, out);
    (void)in_sizes; (void)n_in; (void)out_size;
}
